// TalosJEPA_46677704573588
// MI455X (gfx1250) — matrix operand order repaired and checked by reading, not compiled, not run on hardware
//
#include <hip/hip_runtime.h>
#include <hip/hip_bf16.h>

typedef __attribute__((ext_vector_type(16))) __bf16 v16bf;
typedef __attribute__((ext_vector_type(8)))  float  v8f;

union ABFrag { v16bf v; uint4 q[2]; };

__device__ __forceinline__ __bf16 to_bf16(float f) { return (__bf16)f; }

// Shared epilogue (C 16x16 layout: lane&15 = N, VGPR r = M=r, +8 for hi lanes)
// EPI 0: outf = c                                   (pred_z store)
// EPI 1: outb = bf16(sigmoid(c+v0[col]) * xres * v1[col])   (liquid gate -> s)
// EPI 2: outf = c + v0[col] + xres                  (residual y, pre-LN)
// EPI 3: outb = bf16(v0[row*4 + col/16] * gelu(c))  (MoLE masked hidden)
template<int EPI>
__device__ __forceinline__
void epilogue_elem(float c, int row, int col, int N,
                   const float* __restrict__ v0, const float* __restrict__ v1,
                   const float* __restrict__ xres,
                   float* __restrict__ outf, __bf16* __restrict__ outb)
{
    const size_t idx = (size_t)row * N + col;
    if (EPI == 0) {
        outf[idx] = c;
    } else if (EPI == 1) {
        float gate = 1.0f / (1.0f + __expf(-(c + v0[col])));
        outb[idx] = to_bf16(gate * xres[idx] * v1[col]);
    } else if (EPI == 2) {
        outf[idx] = c + v0[col] + xres[idx];
    } else if (EPI == 3) {
        float h = 0.5f * c * (1.0f + erff(c * 0.70710678118654752f));
        outb[idx] = to_bf16(v0[(size_t)row * 4 + (col >> 4)] * h);
    }
}

// ---------------------------------------------------------------------------
// LDS double-buffered WMMA GEMM for the big D x D matmuls.
// C(Mx1024) = A(Mx1024) * B(1024x1024), row-major bf16, f32 accumulate.
// Block = 256 threads = 8 waves (4 M-waves x 2 N-waves), 128x128 block tile,
// K-step 32, double-buffered LDS tiles, ONE barrier per K-step:
//   g2r(next) -> compute(cur) -> r2l(other) -> barrier
// Buffer indices are compile-time constants (manual unroll-by-2) and the last
// iteration is peeled so the steady-state loop body is branch-free (no
// speculated/duplicated global loads).
// Fragment layouts per CDNA5 ISA 7.12.2:
//   A 16x32 : lanes 0-15 = M, VGPR halves = K (k+0..7 / k+16..23; +8 hi lanes)
//   B 32x16 : lanes 0-15 = N, elements = K in the same k order as A (+8 hi lanes)
// ---------------------------------------------------------------------------
#define LDS_A_STRIDE 40    // 32 K-elems + 8 pad (bf16)
#define LDS_B_STRIDE 132   // 128 N-elems + 4 pad (bf16)

template<int EPI>
__global__ __launch_bounds__(256)
void gemm1024_lds(const __bf16* __restrict__ A, const __bf16* __restrict__ B,
                  const float* __restrict__ v0, const float* __restrict__ v1,
                  const float* __restrict__ xres,
                  float* __restrict__ outf, __bf16* __restrict__ outb)
{
    __shared__ __bf16 As[2][128 * LDS_A_STRIDE];
    __shared__ __bf16 Bs[2][32 * LDS_B_STRIDE];

    const int tid  = threadIdx.x;
    const int wave = tid >> 5, lane = tid & 31;
    const int lh = lane >> 4, lm = lane & 15;
    const int m0 = (wave & 3) * 32;        // wave tile inside block tile
    const int n0 = (wave >> 2) * 64;
    const int m_blk = blockIdx.x * 128;
    const int n_blk = blockIdx.y * 128;

    // global->reg staging map: A row=tid/2, 16-col half=tid&1 (8 KB tile / 32 B per thread)
    //                          B row=tid/8, 16-col chunk=tid&7
    const int ar = tid >> 1, ah = tid & 1;
    const int br = tid >> 3, bc = tid & 7;
    const __bf16* gA = A + (size_t)(m_blk + ar) * 1024 + ah * 16;
    const __bf16* gB = B + (size_t)br * 1024 + n_blk + bc * 16;

    uint4 ra0, ra1, rb0, rb1;
    auto g2r = [&](int kk) {
        const __bf16* pa = gA + kk;
        ra0 = *(const uint4*)pa;       ra1 = *(const uint4*)(pa + 8);
        const __bf16* pb = gB + (size_t)kk * 1024;
        rb0 = *(const uint4*)pb;       rb1 = *(const uint4*)(pb + 8);
    };
    auto r2l = [&](int buf) {
        __bf16* da = &As[buf][ar * LDS_A_STRIDE + ah * 16];
        *(uint4*)da = ra0;             *(uint4*)(da + 8) = ra1;
        __bf16* db = &Bs[buf][br * LDS_B_STRIDE + bc * 16];
        *(uint4*)db = rb0;             *(uint4*)(db + 8) = rb1;
    };

    v8f acc[2][4];
#pragma unroll
    for (int i = 0; i < 2; ++i)
#pragma unroll
        for (int j = 0; j < 4; ++j) acc[i][j] = (v8f)0.0f;

    auto compute = [&](int buf) {
        ABFrag a[2], b[4];
#pragma unroll
        for (int i = 0; i < 2; ++i) {
            const __bf16* ap = &As[buf][(m0 + 16 * i + lm) * LDS_A_STRIDE + lh * 8];
            a[i].q[0] = *(const uint4*)ap;
            a[i].q[1] = *(const uint4*)(ap + 16);
        }
#pragma unroll
        for (int j = 0; j < 4; ++j) {
            const __bf16* bp = &Bs[buf][n0 + 16 * j + lm];
#pragma unroll
            for (int s = 0; s < 16; ++s)
                b[j].v[s] = bp[((s < 8) ? (lh * 8 + s) : (16 + lh * 8 + s - 8)) * LDS_B_STRIDE];
        }
#pragma unroll
        for (int i = 0; i < 2; ++i)
#pragma unroll
            for (int j = 0; j < 4; ++j)
                acc[i][j] = __builtin_amdgcn_wmma_f32_16x16x32_bf16(
                    false, a[i].v, false, b[j].v, (short)0, acc[i][j], false, false);
    };

    g2r(0);
    r2l(0);
    __syncthreads();

    // steady state: 15 branch-free double-iterations (tiles 0..29)
#pragma unroll 1
    for (int kt = 0; kt < 30; kt += 2) {
        // phase 0: compute buf 0 (tile kt), stage tile kt+1 into buf 1
        g2r((kt + 1) * 32);
        compute(0);
        r2l(1);
        __syncthreads();
        // phase 1: compute buf 1 (tile kt+1), stage tile kt+2 into buf 0
        g2r((kt + 2) * 32);
        compute(1);
        r2l(0);
        __syncthreads();
    }
    // peeled tail: tiles 30 (in buf 0) and 31
    g2r(31 * 32);
    compute(0);
    r2l(1);
    __syncthreads();
    compute(1);

#pragma unroll
    for (int i = 0; i < 2; ++i)
#pragma unroll
        for (int j = 0; j < 4; ++j)
#pragma unroll
            for (int r = 0; r < 8; ++r)
                epilogue_elem<EPI>(acc[i][j][r],
                                   m_blk + m0 + 16 * i + lh * 8 + r,
                                   n_blk + n0 + 16 * j + lm,
                                   1024, v0, v1, xres, outf, outb);
}

// ---------------------------------------------------------------------------
// Direct-global WMMA GEMM for the small MoLE matmuls (N=64 or K=64, ~1% FLOPs)
// ---------------------------------------------------------------------------
template<int WAVES_N, int EPI>
__global__ __launch_bounds__(256)
void gemm_bf16_wmma(const __bf16* __restrict__ A, const __bf16* __restrict__ B,
                    int M, int N, int K,
                    const float* __restrict__ v0, const float* __restrict__ v1,
                    const float* __restrict__ xres,
                    float* __restrict__ outf, __bf16* __restrict__ outb)
{
    constexpr int WAVES_M = 8 / WAVES_N;
    const int wave = threadIdx.x >> 5;
    const int lane = threadIdx.x & 31;
    const int lh = lane >> 4;
    const int lm = lane & 15;
    const int m0 = blockIdx.x * (WAVES_M * 32) + (wave % WAVES_M) * 32;
    const int n0 = blockIdx.y * (WAVES_N * 64) + (wave / WAVES_M) * 64;

    v8f acc[2][4];
#pragma unroll
    for (int i = 0; i < 2; ++i)
#pragma unroll
        for (int j = 0; j < 4; ++j) acc[i][j] = (v8f)0.0f;

    for (int kk = 0; kk < K; kk += 32) {
        ABFrag a[2], b[4];
#pragma unroll
        for (int i = 0; i < 2; ++i) {
            const __bf16* ap = A + (size_t)(m0 + 16 * i + lm) * K + (kk + lh * 8);
            a[i].q[0] = *(const uint4*)ap;
            a[i].q[1] = *(const uint4*)(ap + 16);
        }
#pragma unroll
        for (int j = 0; j < 4; ++j) {
            const __bf16* bp = B + (size_t)kk * N + (n0 + 16 * j + lm);
#pragma unroll
            for (int s = 0; s < 16; ++s)
                b[j].v[s] = bp[(size_t)((s < 8) ? (lh * 8 + s) : (16 + lh * 8 + s - 8)) * N];
        }
#pragma unroll
        for (int i = 0; i < 2; ++i)
#pragma unroll
            for (int j = 0; j < 4; ++j)
                acc[i][j] = __builtin_amdgcn_wmma_f32_16x16x32_bf16(
                    false, a[i].v, false, b[j].v, (short)0, acc[i][j], false, false);
    }

#pragma unroll
    for (int i = 0; i < 2; ++i)
#pragma unroll
        for (int j = 0; j < 4; ++j)
#pragma unroll
            for (int r = 0; r < 8; ++r)
                epilogue_elem<EPI>(acc[i][j][r],
                                   m0 + 16 * i + lh * 8 + r,
                                   n0 + 16 * j + lm,
                                   N, v0, v1, xres, outf, outb);
}

// ------------------------- LayerNorm (one wave / token) ---------------------
__global__ __launch_bounds__(256)
void ln_kernel(const float* __restrict__ y, const float* __restrict__ g,
               const float* __restrict__ beta, float* __restrict__ xf,
               __bf16* __restrict__ xb, int T)
{
    const int tok = (int)((blockIdx.x * blockDim.x + threadIdx.x) >> 5);
    const int lane = threadIdx.x & 31;
    if (tok >= T) return;
    const float* yr = y + (size_t)tok * 1024;
    float4 v[8];
    float s = 0.f;
#pragma unroll
    for (int it = 0; it < 8; ++it) {
        v[it] = *(const float4*)(yr + it * 128 + lane * 4);
        s += v[it].x + v[it].y + v[it].z + v[it].w;
    }
    for (int off = 16; off; off >>= 1) s += __shfl_xor(s, off, 32);
    const float mu = s * (1.0f / 1024.0f);
    float sq = 0.f;
#pragma unroll
    for (int it = 0; it < 8; ++it) {
        float a = v[it].x - mu, b = v[it].y - mu, c = v[it].z - mu, d = v[it].w - mu;
        sq += a * a + b * b + c * c + d * d;
    }
    for (int off = 16; off; off >>= 1) sq += __shfl_xor(sq, off, 32);
    const float inv = rsqrtf(sq * (1.0f / 1024.0f) + 1e-5f);
#pragma unroll
    for (int it = 0; it < 8; ++it) {
        const int d0 = it * 128 + lane * 4;
        float o0 = (v[it].x - mu) * inv * g[d0 + 0] + beta[d0 + 0];
        float o1 = (v[it].y - mu) * inv * g[d0 + 1] + beta[d0 + 1];
        float o2 = (v[it].z - mu) * inv * g[d0 + 2] + beta[d0 + 2];
        float o3 = (v[it].w - mu) * inv * g[d0 + 3] + beta[d0 + 3];
        *(float4*)(xf + (size_t)tok * 1024 + d0) = make_float4(o0, o1, o2, o3);
        if (xb) {
            __bf16* xo = xb + (size_t)tok * 1024 + d0;
            xo[0] = to_bf16(o0); xo[1] = to_bf16(o1);
            xo[2] = to_bf16(o2); xo[3] = to_bf16(o3);
        }
    }
}

// -------------------- gating: softmax + top-2 mask (one wave / token) -------
__global__ __launch_bounds__(256)
void gate_kernel(const float* __restrict__ z, const float* __restrict__ gw,
                 const float* __restrict__ gb, float* __restrict__ probs_out,
                 float* __restrict__ mask_out, int T)
{
    const int tok = (int)((blockIdx.x * blockDim.x + threadIdx.x) >> 5);
    const int lane = threadIdx.x & 31;
    if (tok >= T) return;
    const float* zr = z + (size_t)tok * 1024;
    float a0 = 0.f, a1 = 0.f, a2 = 0.f, a3 = 0.f;
    for (int d = lane; d < 1024; d += 32) {
        const float xv = zr[d];
        const float4 w = *(const float4*)(gw + d * 4);   // gate_W is (D,4) row-major
        a0 += xv * w.x; a1 += xv * w.y; a2 += xv * w.z; a3 += xv * w.w;
    }
    for (int off = 16; off; off >>= 1) {
        a0 += __shfl_xor(a0, off, 32); a1 += __shfl_xor(a1, off, 32);
        a2 += __shfl_xor(a2, off, 32); a3 += __shfl_xor(a3, off, 32);
    }
    float p[4] = { a0 + gb[0], a1 + gb[1], a2 + gb[2], a3 + gb[3] };
    const float mx = fmaxf(fmaxf(p[0], p[1]), fmaxf(p[2], p[3]));
    float ssum = 0.f;
#pragma unroll
    for (int e = 0; e < 4; ++e) { p[e] = __expf(p[e] - mx); ssum += p[e]; }
    const float inv = 1.0f / ssum;
#pragma unroll
    for (int e = 0; e < 4; ++e) p[e] *= inv;
    // top-2, lowest index wins ties (matches lax.top_k stability)
    int i1 = 0;
#pragma unroll
    for (int e = 1; e < 4; ++e) if (p[e] > p[i1]) i1 = e;
    int i2 = -1; float b2 = -1.0f;
#pragma unroll
    for (int e = 0; e < 4; ++e) if (e != i1 && p[e] > b2) { b2 = p[e]; i2 = e; }
    if (lane == 0) {
        float* po = probs_out + (size_t)tok * 4;
        float* mo = mask_out + (size_t)tok * 4;
#pragma unroll
        for (int e = 0; e < 4; ++e) {
            po[e] = p[e];
            mo[e] = (e == i1 || e == i2) ? 1.0f : 0.0f;
        }
    }
}

// --------------------------- conversions ------------------------------------
__global__ __launch_bounds__(256)
void cvt4_kernel(const float* __restrict__ in, __bf16* __restrict__ out, int n4)
{
    const int i = blockIdx.x * blockDim.x + threadIdx.x;
    if (i >= n4) return;
    const float4 v = *(const float4*)(in + (size_t)i * 4);
    __bf16* o = out + (size_t)i * 4;
    o[0] = to_bf16(v.x); o[1] = to_bf16(v.y); o[2] = to_bf16(v.z); o[3] = to_bf16(v.w);
}

__global__ __launch_bounds__(256)
void loraA_permute_kernel(const float* __restrict__ a /*E,D,R*/, __bf16* __restrict__ out /*D x 64*/)
{
    const int i = blockIdx.x * blockDim.x + threadIdx.x;   // over D*64
    if (i >= 1024 * 64) return;
    const int d = i >> 6, c = i & 63, e = c >> 4, r = c & 15;
    out[i] = to_bf16(a[((size_t)e * 1024 + d) * 16 + r]);
}

// ----------------------------------------------------------------------------
extern "C" void kernel_launch(void* const* d_in, const int* in_sizes, int n_in,
                              void* d_out, int out_size, void* d_ws, size_t ws_size,
                              hipStream_t stream)
{
    (void)in_sizes; (void)n_in; (void)out_size; (void)ws_size;
    const int T = 8 * 2048, D = 1024;

    const float* x_context = (const float*)d_in[0];
    const float* x_target  = (const float*)d_in[1];
    const float* enc_p[7]  = { (const float*)d_in[2], (const float*)d_in[3], (const float*)d_in[4],
                               (const float*)d_in[5], (const float*)d_in[6], (const float*)d_in[7],
                               (const float*)d_in[8] };
    const float* tgt_p[7]  = { (const float*)d_in[9],  (const float*)d_in[10], (const float*)d_in[11],
                               (const float*)d_in[12], (const float*)d_in[13], (const float*)d_in[14],
                               (const float*)d_in[15] };
    const float* gate_W = (const float*)d_in[16];
    const float* gate_b = (const float*)d_in[17];
    const float* lora_A = (const float*)d_in[18];
    const float* lora_B = (const float*)d_in[19];

    float* out_pred  = (float*)d_out;
    float* out_probs = out_pred + (size_t)T * D;
    float* out_ztgt  = out_probs + (size_t)T * 4;

    // workspace carve-out
    char* w = (char*)d_ws;
    auto carve = [&](size_t bytes) { void* p = (void*)w; w += (bytes + 255) & ~(size_t)255; return p; };
    float*  ws_x    = (float*) carve((size_t)T * D * 4);
    float*  ws_y    = (float*) carve((size_t)T * D * 4);
    __bf16* ws_xb   = (__bf16*)carve((size_t)T * D * 2);
    __bf16* ws_s    = (__bf16*)carve((size_t)T * D * 2);
    __bf16* ws_w    = (__bf16*)carve((size_t)D * D * 2);
    float*  ws_mask = (float*) carve((size_t)T * 4 * 4);
    __bf16* ws_hm   = (__bf16*)carve((size_t)T * 64 * 2);
    __bf16* ws_lA   = (__bf16*)carve((size_t)D * 64 * 2);
    __bf16* ws_lB   = (__bf16*)carve((size_t)64 * D * 2);

    auto cvt = [&](const float* in, __bf16* out, int n) {
        const int n4 = n >> 2;
        cvt4_kernel<<<(n4 + 255) / 256, 256, 0, stream>>>(in, out, n4);
    };

    const dim3 gBig(T / 128, D / 128);   // 128x128 block tiles
    const dim3 gLora(T / 256, 1);        // 256x64 block tile (WAVES_N=1)

    // one liquid stack (3 layers): gemm1(+gate epi) -> gemm2(+resid epi) -> LN
    auto run_stack = [&](const float* xin, const float* const p[7],
                         float* final_f32, __bf16* final_bf16) {
        const float* Wi = p[0]; const float* bi = p[1]; const float* td = p[2];
        const float* Wo = p[3]; const float* bo = p[4]; const float* g  = p[5];
        const float* beta = p[6];
        const float* xf = xin;
        cvt(xin, ws_xb, T * D);
        for (int l = 0; l < 3; ++l) {
            cvt(Wi + (size_t)l * D * D, ws_w, D * D);
            gemm1024_lds<1><<<gBig, 256, 0, stream>>>(
                ws_xb, ws_w, bi + l * D, td + l * D, xf, nullptr, ws_s);
            cvt(Wo + (size_t)l * D * D, ws_w, D * D);
            gemm1024_lds<2><<<gBig, 256, 0, stream>>>(
                ws_s, ws_w, bo + l * D, nullptr, xf, ws_y, nullptr);
            float*  oxf = (l == 2) ? final_f32  : ws_x;
            __bf16* oxb = (l == 2) ? final_bf16 : ws_xb;
            ln_kernel<<<T / 8, 256, 0, stream>>>(ws_y, g + l * D, beta + l * D, oxf, oxb, T);
            xf = oxf;
        }
    };

    // encoder stack -> z_context in ws_x (f32) + ws_xb (bf16)
    run_stack(x_context, enc_p, ws_x, ws_xb);

    // MoLE on z_context
    gate_kernel<<<T / 8, 256, 0, stream>>>(ws_x, gate_W, gate_b, out_probs, ws_mask, T);
    loraA_permute_kernel<<<(1024 * 64 + 255) / 256, 256, 0, stream>>>(lora_A, ws_lA);
    gemm_bf16_wmma<1, 3><<<gLora, 256, 0, stream>>>(
        ws_xb, ws_lA, T, 64, D, ws_mask, nullptr, nullptr, nullptr, ws_hm);
    cvt(lora_B, ws_lB, 64 * D);                       // (E,R,D) flat == (64,1024)
    gemm_bf16_wmma<2, 0><<<gBig, 256, 0, stream>>>(
        ws_hm, ws_lB, T, D, 64, nullptr, nullptr, nullptr, out_pred, nullptr);

    // target stack -> z_target straight into d_out (no bf16 copy needed)
    run_stack(x_target, tgt_p, out_ztgt, nullptr);
}
